// Morgan_87540023427784
// MI455X (gfx1250) — matrix operand order repaired and checked by reading, not compiled, not run on hardware
//
#include <hip/hip_runtime.h>
#include <hip/hip_bf16.h>
#include <math.h>

// ---------------------------------------------------------------------------
// Problem constants (from reference)
// ---------------------------------------------------------------------------
#define NN    8192
#define KK    200
#define KPAD  224      // K padded to multiple of 32 for WMMA K-chunks
#define MPAD  208      // K padded to multiple of 16 for WMMA M-tiles (GEMM1 rows)
#define CIN   512
#define HID   256
#define COUT  40
#define EE    5
#define MH    64
#define KG    40
#define EPS_  1e-5f

typedef _Float16 half_t;
typedef __attribute__((ext_vector_type(16))) _Float16 v16h;
typedef __attribute__((ext_vector_type(8)))  _Float16 v8h;
typedef __attribute__((ext_vector_type(8)))  float    v8f;

union AReg { v16h v; v8h h[2]; };

// ---------------------------------------------------------------------------
// Zero-fill accumulators (graph-replay safe; ws is poisoned, never re-zeroed)
// ---------------------------------------------------------------------------
__global__ void zero_kernel(float* __restrict__ p, int n) {
    int i = blockIdx.x * blockDim.x + threadIdx.x;
    if (i < n) p[i] = 0.f;
}

// ---------------------------------------------------------------------------
// Prep: f16 transposed copies so every WMMA operand load is contiguous b128
// ---------------------------------------------------------------------------
__global__ void prep_ut(const float* __restrict__ U, half_t* __restrict__ Ut) {
    int idx = blockIdx.x * blockDim.x + threadIdx.x;   // over MPAD*NN
    if (idx >= MPAD * NN) return;
    int k = idx / NN, n = idx % NN;                    // Ut[k][n] = U[n][k]
    float v = (k < KK) ? U[(size_t)n * KK + k] : 0.f;
    Ut[idx] = (half_t)v;
}

__global__ void prep_xt(const float* __restrict__ X, half_t* __restrict__ Xt) {
    int idx = blockIdx.x * blockDim.x + threadIdx.x;   // over CIN*NN
    if (idx >= CIN * NN) return;
    int c = idx / NN, n = idx % NN;                    // Xt[c][n] = X[n][c]
    Xt[idx] = (half_t)X[(size_t)n * CIN + c];
}

__global__ void prep_ew2t(const float* __restrict__ eW2, half_t* __restrict__ eW2t) {
    int idx = blockIdx.x * blockDim.x + threadIdx.x;   // over EE*MH*MH
    if (idx >= EE * MH * MH) return;
    int e = idx / (MH * MH), r = idx % (MH * MH);
    int g = r / MH, j = r % MH;                        // eW2t[e][g][j] = eW2[e][j][g]
    eW2t[idx] = (half_t)eW2[((size_t)e * MH + j) * MH + g];
}

// ---------------------------------------------------------------------------
// Stage A: expert tiny-MLPs via WMMA.  One block = one eigen-column slice.
// grid (KK, 4), block 256 (8 waves). Each wave: 16 rows/iter, 16 iters.
// h1 tile (16x64 f16) staged in LDS (144B padded row stride -> conflict-free,
// 16B-aligned b128 reads, k = 16h + s within each lane half h).
// ---------------------------------------------------------------------------
__global__ __launch_bounds__(256) void expert_kernel(
    const float* __restrict__ U,   const float* __restrict__ eW1,
    const float* __restrict__ eb1, const half_t* __restrict__ eW2t,
    const float* __restrict__ eb2, const float* __restrict__ eW3,
    float* __restrict__ Vsum)
{
    __shared__ half_t lds[8][16 * 72];   // 72-half (144B) row stride
    __shared__ float  wsum[8];

    const int k   = blockIdx.x;          // eigen-column 0..199
    const int e   = k / KG;
    const int tid = threadIdx.x;
    const int w   = tid >> 5;
    const int l   = tid & 31;
    const int row = l & 15;
    const int jbase = (l < 16) ? 0 : 32; // which half of h1 this lane computes

    // hoist first-layer weights for this lane's 32 h1 lanes
    float w1r[32], b1r[32];
#pragma unroll
    for (int i = 0; i < 32; ++i) {
        w1r[i] = eW1[e * MH + jbase + i];
        b1r[i] = eb1[e * MH + jbase + i];
    }
    // epilogue constants: this lane's columns g = 16t + (l%16)
    float b2c[4], w3c[4];
#pragma unroll
    for (int t = 0; t < 4; ++t) {
        int g = t * 16 + (l & 15);
        b2c[t] = eb2[e * MH + g];
        w3c[t] = eW3[e * MH + g];
    }

    float vacc = 0.f;
    const int n_start = blockIdx.y * (NN / 4);   // 2048-row slice

    for (int it = 0; it < 16; ++it) {
        const int n0 = n_start + it * 128 + w * 16;

        // h1 = relu(u*eW1 + eb1), f16, into LDS  (lane covers row, 32 j's)
        const float u = U[(size_t)(n0 + row) * KK + k];
        half_t* base = &lds[w][row * 72];
#pragma unroll
        for (int i = 0; i < 4; ++i) {
            v8h p;
#pragma unroll
            for (int q = 0; q < 8; ++q) {
                float h = fmaf(u, w1r[i * 8 + q], b1r[i * 8 + q]);
                p[q] = (half_t)fmaxf(h, 0.f);
            }
            *(v8h*)(base + jbase + i * 8) = p;
        }
        __syncthreads();

        // h2 pre-activation = h1 @ eW2   (8 x v_wmma_f32_16x16x32_f16)
        v8f acc[4] = {};
        const half_t* rb  = &lds[w][row * 72];
        const int     sel = (l < 16) ? 0 : 16;
#pragma unroll
        for (int c = 0; c < 2; ++c) {
            AReg A;
            A.h[0] = *(const v8h*)(rb + 32 * c + sel);       // K  0..7 / 16..23
            A.h[1] = *(const v8h*)(rb + 32 * c + sel + 8);   // K  8..15 /24..31
#pragma unroll
            for (int t = 0; t < 4; ++t) {
                int g  = t * 16 + (l & 15);
                int j0 = 32 * c + ((l < 16) ? 0 : 16);
                v16h B = *(const v16h*)(eW2t + ((size_t)e * MH + g) * MH + j0);
                acc[t] = __builtin_amdgcn_wmma_f32_16x16x32_f16(
                    false, A.v, false, B, (short)0, acc[t], false, false);
            }
        }
        // epilogue: v = sum_g relu(h2+eb2)*eW3 ; accumulate over rows
#pragma unroll
        for (int t = 0; t < 4; ++t)
#pragma unroll
            for (int r = 0; r < 8; ++r)
                vacc += fmaxf(acc[t][r] + b2c[t], 0.f) * w3c[t];
        __syncthreads();
    }

    // reduce to one partial sum per block, atomically add to Vsum[k]
#pragma unroll
    for (int o = 16; o > 0; o >>= 1) vacc += __shfl_xor(vacc, o, 32);
    if (l == 0) wsum[w] = vacc;
    __syncthreads();
    if (tid == 0) {
        float tot = 0.f;
        for (int i = 0; i < 8; ++i) tot += wsum[i];
        atomicAdd(&Vsum[k], tot);
    }
}

// ---------------------------------------------------------------------------
// Gate: V = Vsum/N + eb3 ;  g = softmax(relu(stats@gW1+gb1)@gW2+gb2) ;
// s[k] = g[k/KG] * V[k]
// ---------------------------------------------------------------------------
__global__ void gate_kernel(const float* __restrict__ La,
                            const float* __restrict__ gW1, const float* __restrict__ gb1,
                            const float* __restrict__ gW2, const float* __restrict__ gb2,
                            const float* __restrict__ eb3,
                            const float* __restrict__ Vsum, float* __restrict__ s)
{
    __shared__ float gs[EE];
    if (threadIdx.x == 0) {
        float st[EE], h1[EE], h2[EE];
        for (int i = 0; i < EE; ++i) {
            float a = 0.f;
            for (int q = 0; q < KG; ++q) a += La[i * KG + q];
            st[i] = a / (float)KG;
        }
        for (int j = 0; j < EE; ++j) {
            float a = gb1[j];
            for (int i = 0; i < EE; ++i) a += st[i] * gW1[i * EE + j];
            h1[j] = fmaxf(a, 0.f);
        }
        float m = -1e30f;
        for (int j = 0; j < EE; ++j) {
            float a = gb2[j];
            for (int i = 0; i < EE; ++i) a += h1[i] * gW2[i * EE + j];
            h2[j] = a; m = fmaxf(m, a);
        }
        float sum = 0.f;
        for (int j = 0; j < EE; ++j) { h2[j] = __expf(h2[j] - m); sum += h2[j]; }
        for (int j = 0; j < EE; ++j) gs[j] = h2[j] / sum;
    }
    __syncthreads();
    int kk = threadIdx.x;
    if (kk < 256) {
        float v = 0.f;
        if (kk < KK) {
            int e = kk / KG;
            v = gs[e] * (Vsum[kk] / (float)NN + eb3[e]);
        }
        s[kk] = v;
    }
}

// Us[n][kp] = f16(U[n][kp]*s[kp]) with zero padding to KPAD
__global__ void us_prep(const float* __restrict__ U, const float* __restrict__ s,
                        half_t* __restrict__ Us)
{
    int idx = blockIdx.x * blockDim.x + threadIdx.x;   // NN*KPAD
    if (idx >= NN * KPAD) return;
    int n = idx / KPAD, kp = idx % KPAD;
    float v = (kp < KK) ? U[(size_t)n * KK + kp] * s[kp] : 0.f;
    Us[idx] = (half_t)v;
}

// ---------------------------------------------------------------------------
// GEMM1: T[MPAD,CIN] = Ut[MPAD,NN] @ X[NN,CIN]  (both operands pre-transposed
// so loads are contiguous). grid (13, 8, 8): z splits K=8192 into 1024-chunks,
// f32 atomicAdd epilogue. block = 4 waves, wave owns one 16x16 C tile.
// ---------------------------------------------------------------------------
__global__ __launch_bounds__(128) void gemm1(const half_t* __restrict__ Ut,
                                             const half_t* __restrict__ Xt,
                                             float* __restrict__ T)
{
    const int tid = threadIdx.x, w = tid >> 5, l = tid & 31;
    const int mt = blockIdx.x;                 // 0..12
    const int ct = blockIdx.y * 4 + w;         // 0..31
    const int kk_start = blockIdx.z * 1024;
    const int asel = (l < 16) ? 0 : 16;
    const int bsel = (l < 16) ? 0 : 16;
    const half_t* Abase = Ut + (size_t)(mt * 16 + (l & 15)) * NN;
    const half_t* Bbase = Xt + (size_t)(ct * 16 + (l & 15)) * NN;

    v8f acc = {};
    for (int c = 0; c < 32; ++c) {
        int kk0 = kk_start + c * 32;
        AReg A;
        A.h[0] = *(const v8h*)(Abase + kk0 + asel);
        A.h[1] = *(const v8h*)(Abase + kk0 + asel + 8);
        v16h B = *(const v16h*)(Bbase + kk0 + bsel);
        __builtin_prefetch(Abase + kk0 + 256, 0, 0);   // global_prefetch_b8
        __builtin_prefetch(Bbase + kk0 + 256, 0, 0);
        acc = __builtin_amdgcn_wmma_f32_16x16x32_f16(
            false, A.v, false, B, (short)0, acc, false, false);
    }
    const int mofs = mt * 16 + ((l < 16) ? 0 : 8);
    const int cc   = ct * 16 + (l & 15);
#pragma unroll
    for (int r = 0; r < 8; ++r)
        atomicAdd(&T[(size_t)(mofs + r) * CIN + cc], acc[r]);
}

// TWt[c][k] = f16( sum_j T[k][j] * Ww[j][c] ), zero-padded to KPAD
__global__ void tw_kernel(const float* __restrict__ T, const float* __restrict__ Ww,
                          half_t* __restrict__ TWt)
{
    int k = blockIdx.x;        // 0..223
    int c = threadIdx.x;       // 0..255
    float a = 0.f;
    if (k < KK) {
        for (int j = 0; j < CIN; ++j) a = fmaf(T[(size_t)k * CIN + j], Ww[(size_t)j * HID + c], a);
    }
    TWt[(size_t)c * KPAD + k] = (half_t)a;
}

// ---------------------------------------------------------------------------
// GEMM2: hidden[NN,HID] = Us[NN,KPAD] @ TWt(B)[HID,KPAD] + Wb  -> d_out
// grid (128, 16), block 4 waves; wave owns one 16x16 tile, 7 K-chunks.
// ---------------------------------------------------------------------------
__global__ __launch_bounds__(128) void gemm2(const half_t* __restrict__ Us,
                                             const half_t* __restrict__ TWt,
                                             const float* __restrict__ Wb,
                                             float* __restrict__ hidden)
{
    const int tid = threadIdx.x, w = tid >> 5, l = tid & 31;
    const int mt = blockIdx.x * 4 + w;   // 0..511
    const int ct = blockIdx.y;           // 0..15
    const int asel = (l < 16) ? 0 : 16;
    const int bsel = (l < 16) ? 0 : 16;
    const half_t* Abase = Us  + (size_t)(mt * 16 + (l & 15)) * KPAD;
    const half_t* Bbase = TWt + (size_t)(ct * 16 + (l & 15)) * KPAD;

    v8f acc = {};
#pragma unroll
    for (int c = 0; c < 7; ++c) {
        int kk0 = c * 32;
        AReg A;
        A.h[0] = *(const v8h*)(Abase + kk0 + asel);
        A.h[1] = *(const v8h*)(Abase + kk0 + asel + 8);
        v16h B = *(const v16h*)(Bbase + kk0 + bsel);
        acc = __builtin_amdgcn_wmma_f32_16x16x32_f16(
            false, A.v, false, B, (short)0, acc, false, false);
    }
    const int mofs = mt * 16 + ((l < 16) ? 0 : 8);
    const int cc   = ct * 16 + (l & 15);
    const float bias = Wb[cc];
#pragma unroll
    for (int r = 0; r < 8; ++r)
        hidden[(size_t)(mofs + r) * HID + cc] = acc[r] + bias;
}

// ---------------------------------------------------------------------------
// BatchNorm (training-mode batch stats, biased var) + ReLU, then logits
// ---------------------------------------------------------------------------
__global__ void bnstats(const float* __restrict__ hidden,
                        float* __restrict__ bnSum, float* __restrict__ bnSq)
{
    int c  = threadIdx.x;          // 0..255
    int r0 = blockIdx.x * 256;     // 32 blocks x 256 rows
    float s = 0.f, q = 0.f;
    for (int r = 0; r < 256; ++r) {
        float v = hidden[(size_t)(r0 + r) * HID + c];
        s += v; q = fmaf(v, v, q);
    }
    atomicAdd(&bnSum[c], s);
    atomicAdd(&bnSq[c], q);
}

__global__ void bnnorm(const float* __restrict__ hidden,
                       const float* __restrict__ bnSum, const float* __restrict__ bnSq,
                       const float* __restrict__ gamma, const float* __restrict__ beta,
                       float* __restrict__ Hn)
{
    int row = blockIdx.x, c = threadIdx.x;
    float mu  = bnSum[c] / (float)NN;
    float var = bnSq[c] / (float)NN - mu * mu;
    float inv = rsqrtf(var + EPS_);
    float v = hidden[(size_t)row * HID + c];
    float h = (v - mu) * inv * gamma[c] + beta[c];
    Hn[(size_t)row * HID + c] = fmaxf(h, 0.f);
}

// logits = Hn @ Mw + Mb ; log_softmax over 40 cols. One wave per row;
// lane covers col=lane and (lanes 0..7) col=lane+32, wave32 shfl reductions.
__global__ __launch_bounds__(256) void logits_kernel(const float* __restrict__ Hn,
                                                     const float* __restrict__ Mw,
                                                     const float* __restrict__ Mb,
                                                     float* __restrict__ out)
{
    int tid = threadIdx.x, w = tid >> 5, l = tid & 31;
    int row = blockIdx.x * 8 + w;
    const float* h = Hn + (size_t)row * HID;
    float a0 = 0.f, a1 = 0.f;
    int c0 = l, c1 = l + 32;
    for (int c = 0; c < HID; ++c) {
        float hv = h[c];
        a0 = fmaf(hv, Mw[(size_t)c * COUT + c0], a0);
        if (l < 8) a1 = fmaf(hv, Mw[(size_t)c * COUT + c1], a1);
    }
    a0 += Mb[c0];
    if (l < 8) a1 += Mb[c1];
    float m = (l < 8) ? fmaxf(a0, a1) : a0;
#pragma unroll
    for (int o = 16; o > 0; o >>= 1) m = fmaxf(m, __shfl_xor(m, o, 32));
    float se = __expf(a0 - m) + ((l < 8) ? __expf(a1 - m) : 0.f);
#pragma unroll
    for (int o = 16; o > 0; o >>= 1) se += __shfl_xor(se, o, 32);
    float ls = __logf(se);
    out[(size_t)row * COUT + c0] = a0 - m - ls;
    if (l < 8) out[(size_t)row * COUT + c1] = a1 - m - ls;
}

// ---------------------------------------------------------------------------
// Host-side launch
// ---------------------------------------------------------------------------
extern "C" void kernel_launch(void* const* d_in, const int* in_sizes, int n_in,
                              void* d_out, int out_size, void* d_ws, size_t ws_size,
                              hipStream_t stream)
{
    const float* X   = (const float*)d_in[0];
    const float* La  = (const float*)d_in[1];
    const float* U   = (const float*)d_in[2];
    const float* eW1 = (const float*)d_in[3];
    const float* eb1 = (const float*)d_in[4];
    const float* eW2 = (const float*)d_in[5];
    const float* eb2 = (const float*)d_in[6];
    const float* eW3 = (const float*)d_in[7];
    const float* eb3 = (const float*)d_in[8];
    const float* gW1 = (const float*)d_in[9];
    const float* gb1 = (const float*)d_in[10];
    const float* gW2 = (const float*)d_in[11];
    const float* gb2 = (const float*)d_in[12];
    const float* Ww  = (const float*)d_in[13];
    const float* Wb  = (const float*)d_in[14];
    const float* bng = (const float*)d_in[15];
    const float* bnb = (const float*)d_in[16];
    const float* Mw  = (const float*)d_in[17];
    const float* Mb  = (const float*)d_in[18];

    float* out_ls     = (float*)d_out;           // [NN][COUT]
    float* out_hidden = out_ls + (size_t)NN * COUT;  // [NN][HID]

    // workspace bump allocator (256B aligned)
    char* wp = (char*)d_ws;
    auto alloc = [&](size_t bytes) {
        char* p = wp;
        wp += (bytes + 255) & ~(size_t)255;
        return p;
    };
    float*  T     = (float*)alloc((size_t)MPAD * CIN * 4); // accum (zeroed)
    float*  Vsum  = (float*)alloc(256 * 4);                // accum (zeroed)
    float*  bnSum = (float*)alloc(256 * 4);                // accum (zeroed)
    float*  bnSq  = (float*)alloc(256 * 4);                // accum (zeroed)
    float*  s     = (float*)alloc(256 * 4);
    half_t* Ut    = (half_t*)alloc((size_t)MPAD * NN * 2);
    half_t* Xt    = (half_t*)alloc((size_t)CIN * NN * 2);
    half_t* eW2t  = (half_t*)alloc((size_t)EE * MH * MH * 2);
    half_t* Us    = (half_t*)alloc((size_t)NN * KPAD * 2);
    half_t* TWt   = (half_t*)alloc((size_t)HID * KPAD * 2);
    float*  Hn    = (float*)alloc((size_t)NN * HID * 4);
    (void)ws_size; (void)in_sizes; (void)n_in; (void)out_size;

    // T..bnSq are contiguous (all sizes are 256B multiples): zero in one shot
    const int zeroN = MPAD * CIN + 3 * 256;
    zero_kernel<<<(zeroN + 255) / 256, 256, 0, stream>>>(T, zeroN);

    prep_ut  <<<(MPAD * NN + 255) / 256, 256, 0, stream>>>(U, Ut);
    prep_xt  <<<(CIN * NN + 255) / 256, 256, 0, stream>>>(X, Xt);
    prep_ew2t<<<(EE * MH * MH + 255) / 256, 256, 0, stream>>>(eW2, eW2t);

    expert_kernel<<<dim3(KK, 4), 256, 0, stream>>>(U, eW1, eb1, eW2t, eb2, eW3, Vsum);
    gate_kernel  <<<1, 256, 0, stream>>>(La, gW1, gb1, gW2, gb2, eb3, Vsum, s);
    us_prep      <<<(NN * KPAD + 255) / 256, 256, 0, stream>>>(U, s, Us);

    gemm1    <<<dim3(13, 8, 8), 128, 0, stream>>>(Ut, Xt, T);
    tw_kernel<<<KPAD, HID, 0, stream>>>(T, Ww, TWt);
    gemm2    <<<dim3(128, 16), 128, 0, stream>>>(Us, TWt, Wb, out_hidden);

    bnstats<<<32, 256, 0, stream>>>(out_hidden, bnSum, bnSq);
    bnnorm <<<NN, HID, 0, stream>>>(out_hidden, bnSum, bnSq, bng, bnb, Hn);
    logits_kernel<<<NN / 8, 256, 0, stream>>>(Hn, Mw, Mb, out_ls);
}
